// INNModel_67224828117643
// MI455X (gfx1250) — compile-verified, matrix operand order checked by reading, not run on hardware
//
#include <hip/hip_runtime.h>
#include <hip/hip_bf16.h>
#include <stdint.h>

typedef _Float16 h16;
typedef __attribute__((ext_vector_type(16))) _Float16 v16h;
typedef __attribute__((ext_vector_type(8)))  _Float16 v8h;
typedef __attribute__((ext_vector_type(2)))  _Float16 h16x2;
typedef __attribute__((ext_vector_type(8)))  float    v8f;
typedef __attribute__((ext_vector_type(2)))  float    f32x2;

#define EPSV 1e-5f

// ---------------- workspace layout ----------------
// halves region (byte 0):
#define H_CP     0      // couplings 1..4: 4 layers * 4 mlps * 2048 halves (frag-swizzled)
#define H_JL0    32768  // jl0 (+bn1 folded): 4 frags * 512
#define H_JL123  34816  // jl1..3 (+bn2..4 folded): 3 * 8 frags * 512
#define H_CP0    47104  // cp0: 4 mlps * 2048 (K-padded din<=2, dout<=2)
#define H_LIN    55296  // lin: 2 frags * 512 (K-tiles from XA/XB, N-tile with 3 valid)
#define H_TOTAL  56320
#define WS_FLOAT_BYTE_OFF (H_TOTAL*2)
// float region (indices in floats):
#define F_CPB   0      // 4 layers * 4 mlps * 64 (b1[32] | b2pad[32])
#define F_JLB   1024   // 4 * 64 (jl0..jl3 biases, bn-folded, tile-swizzled)
#define F_CP0B  1280   // 4 mlps * 64
#define F_BN0   1536   // a[0..2] @0, shift[0..2] @4
#define F_LINB  1544   // 16 (3 valid)
#define F_CLD   1560   // constant logdet

struct PtrParams {
  const float* bn[5][4];
  const float* cp[5][4][4];
  const float* jl[4][2];
  const float* lin[2];
};

// ---------------- device helpers ----------------
__device__ __forceinline__ void lds_fence() {
  asm volatile("s_wait_dscnt 0" ::: "memory");
}
// f16 A/B layout: lane L, half h -> K = (h<8) ? 8*(L>>4)+h : 8+8*(L>>4)+h
__device__ __forceinline__ int kmap(int h, int hi) {
  return (h < 8) ? (hi * 8 + h) : (8 + hi * 8 + h);
}
// storage permutation: unit u lives at column sig(u); col c holds unit (c&1)*16+(c>>1)
__device__ __forceinline__ int sig(int u) { return 2 * (u & 15) + (u >> 4); }

__device__ __forceinline__ v16h ldA16(const h16* base, int lane) {
  int m = lane & 15, hi = lane >> 4;
  v8h lo = *(const v8h*)(base + m * 32 + hi * 8);
  v8h up = *(const v8h*)(base + m * 32 + 16 + hi * 8);
  return __builtin_shufflevector(lo, up, 0, 1, 2, 3, 4, 5, 6, 7, 8, 9, 10, 11,
                                 12, 13, 14, 15);
}
__device__ __forceinline__ v16h ldBfrag(const h16* frag, int lane) {
  return *(const v16h*)(frag + (size_t)lane * 16);
}
__device__ __forceinline__ v8f wmma32(v16h a, v16h b, v8f c) {
  return __builtin_amdgcn_wmma_f32_16x16x32_f16(false, a, false, b, (short)0, c,
                                                false, false);
}

// st = [tanh?](relu(in @ W1 + b1) @ W2 + b2)
__device__ __forceinline__ void mlp_stage(const h16* in, h16* HS, float* ST,
                                          const h16* __restrict__ wg,
                                          const float* __restrict__ bg,
                                          bool do_tanh, int lane) {
  const int n = lane & 15, hi = lane >> 4;
  v16h a1 = ldA16(in, lane);
  v8f c0 = {}, c1 = {};
  c0 = wmma32(a1, ldBfrag(wg + 0, lane), c0);
  c1 = wmma32(a1, ldBfrag(wg + 512, lane), c1);
  float b0 = bg[n], b1v = bg[16 + n];
#pragma unroll
  for (int r = 0; r < 8; ++r) {
    float v0 = fmaxf(c0[r] + b0, 0.f);
    float v1 = fmaxf(c1[r] + b1v, 0.f);
    *(h16x2*)(HS + (r + 8 * hi) * 32 + 2 * n) = (h16x2){(h16)v0, (h16)v1};
  }
  lds_fence();
  v16h a2 = ldA16(HS, lane);
  v8f d0 = {}, d1 = {};
  d0 = wmma32(a2, ldBfrag(wg + 1024, lane), d0);
  d1 = wmma32(a2, ldBfrag(wg + 1536, lane), d1);
  float e0 = bg[32 + n], e1 = bg[48 + n];
#pragma unroll
  for (int r = 0; r < 8; ++r) {
    float v0 = d0[r] + e0;
    float v1 = d1[r] + e1;
    if (do_tanh) { v0 = tanhf(v0); v1 = tanhf(v1); }
    *(f32x2*)(ST + (r + 8 * hi) * 32 + 2 * n) = (f32x2){v0, v1};
  }
  lds_fence();
}

// jl (50x50, bn-folded): [XA|XB] <- [XA|XB] @ Wt + b  (2 K-tiles, 4 N-tiles paired)
__device__ __forceinline__ void jl_stage(h16* XA, h16* XB,
                                         const h16* __restrict__ wg,
                                         const float* __restrict__ jb, int lane) {
  const int n = lane & 15, hi = lane >> 4;
  v16h a0 = ldA16(XA, lane);
  v16h a1 = ldA16(XB, lane);
#pragma unroll
  for (int p = 0; p < 2; ++p) {
    const int tA = 2 * p, tB = 2 * p + 1;
    v8f cA = {}, cB = {};
    cA = wmma32(a0, ldBfrag(wg + (tA * 2 + 0) * 512, lane), cA);
    cA = wmma32(a1, ldBfrag(wg + (tA * 2 + 1) * 512, lane), cA);
    cB = wmma32(a0, ldBfrag(wg + (tB * 2 + 0) * 512, lane), cB);
    cB = wmma32(a1, ldBfrag(wg + (tB * 2 + 1) * 512, lane), cB);
    float bA = jb[tA * 16 + n], bB = jb[tB * 16 + n];
    h16* dst = p ? XB : XA;
#pragma unroll
    for (int r = 0; r < 8; ++r) {
      *(h16x2*)(dst + (r + 8 * hi) * 32 + 2 * n) =
          (h16x2){(h16)(cA[r] + bA), (h16)(cB[r] + bB)};
    }
  }
  lds_fence();
}

// jl0 (3->50, bn1-folded): single K-tile from XA, 4 N-tiles paired
__device__ __forceinline__ void jl0_stage(h16* XA, h16* XB,
                                          const h16* __restrict__ wg,
                                          const float* __restrict__ jb, int lane) {
  const int n = lane & 15, hi = lane >> 4;
  v16h a0 = ldA16(XA, lane);
#pragma unroll
  for (int p = 0; p < 2; ++p) {
    const int tA = 2 * p, tB = 2 * p + 1;
    v8f cA = {}, cB = {};
    cA = wmma32(a0, ldBfrag(wg + tA * 512, lane), cA);
    cB = wmma32(a0, ldBfrag(wg + tB * 512, lane), cB);
    float bA = jb[tA * 16 + n], bB = jb[tB * 16 + n];
    h16* dst = p ? XB : XA;
#pragma unroll
    for (int r = 0; r < 8; ++r) {
      *(h16x2*)(dst + (r + 8 * hi) * 32 + 2 * n) =
          (h16x2){(h16)(cA[r] + bA), (h16)(cB[r] + bB)};
    }
  }
  lds_fence();
}

// ---------------- prep kernel (1 block, 256 threads) ----------------
__global__ void inn_prep(PtrParams P, void* wsv) {
  h16* wsH = (h16*)wsv;
  float* wsF = (float*)((char*)wsv + WS_FLOAT_BYTE_OFF);
  const int tid = threadIdx.x;
  const int NT = 256;

  // couplings 1..4
  for (int li = 0; li < 4; ++li) {
    for (int m = 0; m < 4; ++m) {
      const float* W1 = P.cp[li + 1][m][0];  // (25,32)
      const float* b1 = P.cp[li + 1][m][1];
      const float* W2 = P.cp[li + 1][m][2];  // (32,25)
      const float* b2 = P.cp[li + 1][m][3];
      h16* base = wsH + H_CP + (li * 4 + m) * 2048;
      for (int idx = tid; idx < 2048; idx += NT) {
        int f = idx >> 9, r = idx & 511;
        int lf = r >> 4, h = r & 15;
        int n = lf & 15, hi = lf >> 4;
        int c = kmap(h, hi);
        int u = (c & 1) * 16 + (c >> 1);
        float v;
        if (f < 2) v = (u < 25) ? W1[u * 32 + (f * 16 + n)] : 0.f;
        else {
          int j = (f - 2) * 16 + n;
          v = (j < 25) ? W2[u * 25 + j] : 0.f;
        }
        base[idx] = (h16)v;
      }
      float* bz = wsF + F_CPB + li * 256 + m * 64;
      for (int idx = tid; idx < 64; idx += NT)
        bz[idx] = (idx < 32) ? b1[idx] : ((idx - 32) < 25 ? b2[idx - 32] : 0.f);
    }
  }
  // cp0: K-padded tiny MLPs
  {
    const int dins[4] = {1, 1, 2, 2};
    const int douts[4] = {2, 2, 1, 1};
    for (int m = 0; m < 4; ++m) {
      const float* W1 = P.cp[0][m][0];
      const float* b1 = P.cp[0][m][1];
      const float* W2 = P.cp[0][m][2];
      const float* b2 = P.cp[0][m][3];
      int din = dins[m], dout = douts[m];
      h16* base = wsH + H_CP0 + m * 2048;
      for (int idx = tid; idx < 2048; idx += NT) {
        int f = idx >> 9, r = idx & 511;
        int lf = r >> 4, h = r & 15;
        int n = lf & 15, hi = lf >> 4;
        int c = kmap(h, hi);
        int u = (c & 1) * 16 + (c >> 1);
        float v;
        if (f < 2) v = (u < din) ? W1[u * 32 + (f * 16 + n)] : 0.f;
        else {
          int j = (f - 2) * 16 + n;
          v = (j < dout) ? W2[u * dout + j] : 0.f;
        }
        base[idx] = (h16)v;
      }
      float* bz = wsF + F_CP0B + m * 64;
      for (int idx = tid; idx < 64; idx += NT)
        bz[idx] = (idx < 32) ? b1[idx] : ((idx - 32) < dout ? b2[idx - 32] : 0.f);
    }
  }
  // jl0 (3->50) with bn1 folded
  {
    const float* W = P.jl[0][0];  // (50,3)
    const float* b = P.jl[0][1];
    const float* g = P.bn[1][0];
    const float* be = P.bn[1][1];
    const float* mu = P.bn[1][2];
    const float* va = P.bn[1][3];
    h16* base = wsH + H_JL0;
    for (int idx = tid; idx < 2048; idx += NT) {
      int t = idx >> 9, r = idx & 511;
      int lf = r >> 4, h = r & 15;
      int n = lf & 15, hi = lf >> 4;
      int c = kmap(h, hi);
      int u = (c & 1) * 16 + (c >> 1);
      int lo = (t & 1) ? 16 + n : n;
      int realout = (t >> 1) * 25 + lo;
      float v = 0.f;
      if (u < 3 && lo < 25) {
        float a = g[realout] * rsqrtf(va[realout] + EPSV);
        v = W[realout * 3 + u] * a;
      }
      base[idx] = (h16)v;
    }
    float* jb = wsF + F_JLB;
    for (int idx = tid; idx < 64; idx += NT) {
      int t = idx >> 4, n = idx & 15;
      int lo = (t & 1) ? 16 + n : n;
      int realout = (t >> 1) * 25 + lo;
      float v = 0.f;
      if (lo < 25) {
        float a = g[realout] * rsqrtf(va[realout] + EPSV);
        v = b[realout] * a + (be[realout] - mu[realout] * a);
      }
      jb[idx] = v;
    }
  }
  // jl1..3 (50x50) with bn2..4 folded
  for (int li = 1; li < 4; ++li) {
    const float* W = P.jl[li][0];  // (50,50)
    const float* b = P.jl[li][1];
    const float* g = P.bn[li + 1][0];
    const float* be = P.bn[li + 1][1];
    const float* mu = P.bn[li + 1][2];
    const float* va = P.bn[li + 1][3];
    h16* base = wsH + H_JL123 + (li - 1) * 4096;
    for (int idx = tid; idx < 4096; idx += NT) {
      int f = idx >> 9, r = idx & 511;
      int t = f >> 1, kk = f & 1;
      int lf = r >> 4, h = r & 15;
      int n = lf & 15, hi = lf >> 4;
      int c = kmap(h, hi);
      int u = (c & 1) * 16 + (c >> 1);
      int lo = (t & 1) ? 16 + n : n;
      int realin = kk * 25 + u;
      int realout = (t >> 1) * 25 + lo;
      float v = 0.f;
      if (u < 25 && lo < 25) {
        float a = g[realout] * rsqrtf(va[realout] + EPSV);
        v = W[realout * 50 + realin] * a;
      }
      base[idx] = (h16)v;
    }
    float* jb = wsF + F_JLB + li * 64;
    for (int idx = tid; idx < 64; idx += NT) {
      int t = idx >> 4, n = idx & 15;
      int lo = (t & 1) ? 16 + n : n;
      int realout = (t >> 1) * 25 + lo;
      float v = 0.f;
      if (lo < 25) {
        float a = g[realout] * rsqrtf(va[realout] + EPSV);
        v = b[realout] * a + (be[realout] - mu[realout] * a);
      }
      jb[idx] = v;
    }
  }
  // lin (3,50): 2 K-tile frags, single N tile
  {
    const float* LW = P.lin[0];
    const float* lb = P.lin[1];
    h16* base = wsH + H_LIN;
    for (int idx = tid; idx < 1024; idx += NT) {
      int kk = idx >> 9, r = idx & 511;
      int lf = r >> 4, h = r & 15;
      int n = lf & 15, hi = lf >> 4;
      int c = kmap(h, hi);
      int u = (c & 1) * 16 + (c >> 1);
      int realin = kk * 25 + u;
      float v = (u < 25 && n < 3) ? LW[n * 50 + realin] : 0.f;
      base[idx] = (h16)v;
    }
    for (int idx = tid; idx < 16; idx += NT)
      (wsF + F_LINB)[idx] = (idx < 3) ? lb[idx] : 0.f;
  }
  // bn0 affine
  for (int c = tid; c < 3; c += NT) {
    float a = P.bn[0][0][c] * rsqrtf(P.bn[0][3][c] + EPSV);
    wsF[F_BN0 + c] = a;
    wsF[F_BN0 + 4 + c] = P.bn[0][1][c] - P.bn[0][2][c] * a;
  }

  // constant logdet: bn terms + 0.5*logdet(W^T W)
  __shared__ float M[2500];
  __shared__ float cldS;
  if (tid == 0) {
    float c = 0.f;
    for (int i = 0; i < 5; ++i) {
      int d = (i == 0) ? 3 : 50;
      for (int k = 0; k < d; ++k)
        c += logf(fabsf(P.bn[i][0][k])) - 0.5f * logf(P.bn[i][3][k] + EPSV);
    }
    const float* W = P.jl[0][0];
    float m3[9];
    for (int r = 0; r < 3; ++r)
      for (int cc = 0; cc < 3; ++cc) {
        float a = 0.f;
        for (int k = 0; k < 50; ++k) a += W[k * 3 + r] * W[k * 3 + cc];
        m3[r * 3 + cc] = a;
      }
    float det = m3[0] * (m3[4] * m3[8] - m3[5] * m3[7])
              - m3[1] * (m3[3] * m3[8] - m3[5] * m3[6])
              + m3[2] * (m3[3] * m3[7] - m3[4] * m3[6]);
    c += 0.5f * logf(fabsf(det));
    cldS = c;
  }
  __syncthreads();
  for (int li = 0; li < 3; ++li) {
    const float* W = P.jl[li + 1][0];
    for (int idx = tid; idx < 2500; idx += NT) {
      int r = idx / 50, cc = idx % 50;
      float a = 0.f;
      for (int k = 0; k < 50; ++k) a += W[k * 50 + r] * W[k * 50 + cc];
      M[idx] = a;
    }
    __syncthreads();
    for (int j = 0; j < 49; ++j) {
      float piv = M[j * 50 + j];
      for (int ii = j + 1 + tid; ii < 50; ii += NT) M[ii * 50 + j] /= piv;
      __syncthreads();
      int rem = 49 - j;
      if (rem > 0) {
        for (int idx = tid; idx < rem * rem; idx += NT) {
          int ii = j + 1 + idx / rem, cc = j + 1 + idx % rem;
          M[ii * 50 + cc] -= M[ii * 50 + j] * M[j * 50 + cc];
        }
      }
      __syncthreads();
    }
    if (tid == 0) {
      float s = 0.f;
      for (int j = 0; j < 50; ++j) s += logf(fabsf(M[j * 50 + j]));
      cldS += 0.5f * s;
    }
    __syncthreads();
  }
  if (tid == 0) wsF[F_CLD] = cldS;
}

// ---------------- main kernel: 128 threads = 4 independent waves ----------------
__global__ void __launch_bounds__(128)
inn_forward(const float* __restrict__ sphere, const void* __restrict__ wsv,
            float* __restrict__ out, int N) {
  const h16* wsH = (const h16*)wsv;
  const float* wsF = (const float*)((const char*)wsv + WS_FLOAT_BYTE_OFF);

  __shared__ __align__(64) h16 xa[4][16][32];
  __shared__ __align__(64) h16 xb[4][16][32];
  __shared__ __align__(64) h16 hs[4][16][32];
  __shared__ __align__(64) float st[4][2][16][32];

  const int tid = threadIdx.x;
  const int lane = tid & 31;
  const int w = tid >> 5;
  h16* XA = &xa[w][0][0];
  h16* XB = &xb[w][0][0];
  h16* HS = &hs[w][0][0];
  float* ST0 = &st[w][0][0][0];
  float* ST1 = &st[w][1][0][0];
  const int row = lane & 15, half = lane >> 4;
  const int rowbase = blockIdx.x * 64 + w * 16;
  float ld = 0.f;

  // zero xa/xb (each lane clears one 64B row)
  {
    uint32_t* p = (uint32_t*)(half ? (XB + row * 32) : (XA + row * 32));
#pragma unroll
    for (int i = 0; i < 16; ++i) p[i] = 0;
  }
  lds_fence();

  // ---- load + bn0 (3 FMAs) ----
  float xp = 0.f, xq = 0.f, x0keep = 0.f;
  if (lane < 16) {
    int grow = rowbase + lane;
    float x0 = 0.f;
    if (grow < N) {
      x0 = sphere[(size_t)grow * 3 + 0];
      xp = sphere[(size_t)grow * 3 + 1];
      xq = sphere[(size_t)grow * 3 + 2];
    }
    x0 = x0 * wsF[F_BN0 + 0] + wsF[F_BN0 + 4];
    xp = xp * wsF[F_BN0 + 1] + wsF[F_BN0 + 5];
    xq = xq * wsF[F_BN0 + 2] + wsF[F_BN0 + 6];
    x0keep = x0;
    XA[lane * 32 + sig(0)] = (h16)x0;
  }
  lds_fence();

  // ---- cp0 via WMMA ----
  {
    const h16* w0 = wsH + H_CP0;
    const float* b0 = wsF + F_CP0B;
    mlp_stage(XA, HS, ST0, w0 + 0,    b0 + 0,  true,  lane);  // s1
    mlp_stage(XA, HS, ST1, w0 + 2048, b0 + 64, false, lane);  // t1
    if (lane < 16) {
      float s1a = ST0[lane * 32 + sig(0)], s1b = ST0[lane * 32 + sig(1)];
      float t1a = ST1[lane * 32 + sig(0)], t1b = ST1[lane * 32 + sig(1)];
      float y2a = xp * __expf(s1a) + t1a;
      float y2b = xq * __expf(s1b) + t1b;
      ld += s1a + s1b;
      xp = y2a; xq = y2b;  // keep y2 in regs
      XB[lane * 32 + sig(0)] = (h16)y2a;
      XB[lane * 32 + sig(1)] = (h16)y2b;
    }
    lds_fence();
    mlp_stage(XB, HS, ST0, w0 + 4096, b0 + 128, true,  lane);  // s2
    mlp_stage(XB, HS, ST1, w0 + 6144, b0 + 192, false, lane);  // t2
    if (lane < 16) {
      float s2 = ST0[lane * 32 + sig(0)];
      float t2 = ST1[lane * 32 + sig(0)];
      float y1 = x0keep * __expf(s2) + t2;
      ld += s2;
      // stage (y1, y2a, y2b) into XA as jl0's A operand
      XA[lane * 32 + sig(0)] = (h16)y1;
      XA[lane * 32 + sig(1)] = (h16)xp;
      XA[lane * 32 + sig(2)] = (h16)xq;
    }
    lds_fence();
  }

  // ---- jl0 (3->50) + bn1, via WMMA ----
  jl0_stage(XA, XB, wsH + H_JL0, wsF + F_JLB, lane);

  // ---- 4 rounds: WMMA coupling (+ folded-bn jl for i<4) ----
  for (int i = 1; i <= 4; ++i) {
    const h16* wl = wsH + H_CP + (i - 1) * 8192;
    const float* bl = wsF + F_CPB + (i - 1) * 256;

    mlp_stage(XA, HS, ST0, wl + 0,    bl + 0,  true,  lane);  // s1
    mlp_stage(XA, HS, ST1, wl + 2048, bl + 64, false, lane);  // t1
    {  // y2 = x2 * exp(s1) + t1 ; ld += s1
      int j0 = half ? 13 : 0, j1 = half ? 25 : 13;
      for (int j = j0; j < j1; ++j) {
        int o = row * 32 + sig(j);
        float s = ST0[o], t = ST1[o];
        XB[o] = (h16)((float)XB[o] * __expf(s) + t);
        ld += s;
      }
    }
    lds_fence();
    mlp_stage(XB, HS, ST0, wl + 4096, bl + 128, true,  lane);  // s2
    mlp_stage(XB, HS, ST1, wl + 6144, bl + 192, false, lane);  // t2
    {  // y1 = x1 * exp(s2) + t2 ; ld += s2
      int j0 = half ? 13 : 0, j1 = half ? 25 : 13;
      for (int j = j0; j < j1; ++j) {
        int o = row * 32 + sig(j);
        float s = ST0[o], t = ST1[o];
        XA[o] = (h16)((float)XA[o] * __expf(s) + t);
        ld += s;
      }
    }
    lds_fence();

    if (i < 4)
      jl_stage(XA, XB, wsH + H_JL123 + (i - 1) * 4096,
               wsF + F_JLB + i * 64, lane);
  }

  // ---- final 50->3 linear via WMMA ----
  {
    const int n = lane & 15, hi = lane >> 4;
    v16h a0 = ldA16(XA, lane);
    v16h a1 = ldA16(XB, lane);
    v8f c = {};
    c = wmma32(a0, ldBfrag(wsH + H_LIN + 0, lane), c);
    c = wmma32(a1, ldBfrag(wsH + H_LIN + 512, lane), c);
    float b = wsF[F_LINB + n];
#pragma unroll
    for (int r = 0; r < 8; ++r) ST0[(r + 8 * hi) * 32 + n] = c[r] + b;
    lds_fence();
  }

  // combine per-half logdet partials
  float ldtot = ld + __shfl_xor(ld, 16, 32);

  if (lane < 16) {
    int grow = rowbase + lane;
    if (grow < N) {
      out[(size_t)grow * 3 + 0] = tanhf(ST0[lane * 32 + 0]);
      out[(size_t)grow * 3 + 1] = tanhf(ST0[lane * 32 + 1]);
      out[(size_t)grow * 3 + 2] = tanhf(ST0[lane * 32 + 2]);
      out[(size_t)3 * N + grow] = 0.f;                   // logp
      out[(size_t)4 * N + grow] = ldtot + wsF[F_CLD];    // logdet
    }
  }
}

// ---------------- host ----------------
extern "C" void kernel_launch(void* const* d_in, const int* in_sizes, int n_in,
                              void* d_out, int out_size, void* d_ws, size_t ws_size,
                              hipStream_t stream) {
  PtrParams P;
  int p = 1;  // d_in[0] = sphere
  auto nx = [&]() { return (const float*)d_in[p++]; };
  for (int i = 0; i < 5; ++i) {
    for (int j = 0; j < 4; ++j) P.bn[i][j] = nx();
    for (int m = 0; m < 4; ++m)
      for (int k = 0; k < 4; ++k) P.cp[i][m][k] = nx();
    if (i < 4) { P.jl[i][0] = nx(); P.jl[i][1] = nx(); }
  }
  P.lin[0] = nx();
  P.lin[1] = nx();

  const int N = in_sizes[0] / 3;
  hipLaunchKernelGGL(inn_prep, dim3(1), dim3(256), 0, stream, P, d_ws);
  const int blocks = (N + 63) / 64;
  hipLaunchKernelGGL(inn_forward, dim3(blocks), dim3(128), 0, stream,
                     (const float*)d_in[0], (const void*)d_ws, (float*)d_out, N);
}
